// Attention_26938034880689
// MI455X (gfx1250) — matrix operand order repaired and checked by reading, not compiled, not run on hardware
//
#include <hip/hip_runtime.h>
#include <math.h>

#define HEAD_DIM 64
#define NH 32
#define NKV 8
#define SEQ 2048
#define DMODEL 2048
#define SP (SEQ + 32)   // padded key length (zero-filled pad)

typedef __attribute__((ext_vector_type(16))) __bf16 bf16x16;
typedef __attribute__((ext_vector_type(8)))  float  f32x8;
typedef __attribute__((ext_vector_type(4)))  unsigned int u32x4;

union Frag16 {
    bf16x16 v;
    u32x4   q[2];
};

__device__ __forceinline__ unsigned short f2bf(float x) {
    union { float f; unsigned u; } v; v.f = x;
    unsigned r = v.u + 0x7FFFu + ((v.u >> 16) & 1u);  // round-to-nearest-even
    return (unsigned short)(r >> 16);
}
__device__ __forceinline__ float bf2f(unsigned short h) {
    union { unsigned u; float f; } v; v.u = ((unsigned)h) << 16;
    return v.f;
}

// DPP16 lane permutes (VALU pipe, no LDS): butterfly over a 16-lane group.
// quad_perm(1,0,3,2)=xor1 (0xB1), quad_perm(2,3,0,1)=xor2 (0x4E),
// ROW_HALF_MIRROR=xor7 (0x141), ROW_MIRROR=xor15 (0x140); {1,2,7,15} spans 16.
template <int CTRL>
__device__ __forceinline__ float dpp_f(float x) {
    return __int_as_float(__builtin_amdgcn_update_dpp(
        0, __float_as_int(x), CTRL, 0xf, 0xf, true));
}
__device__ __forceinline__ float grp16_max(float x) {
    x = fmaxf(x, dpp_f<0xB1>(x));
    x = fmaxf(x, dpp_f<0x4E>(x));
    x = fmaxf(x, dpp_f<0x141>(x));
    x = fmaxf(x, dpp_f<0x140>(x));
    return x;
}
__device__ __forceinline__ float grp16_sum(float x) {
    x += dpp_f<0xB1>(x);
    x += dpp_f<0x4E>(x);
    x += dpp_f<0x141>(x);
    x += dpp_f<0x140>(x);
    return x;
}

// ---------------------------------------------------------------------------
// One-time precision conversion kernels (bandwidth-bound, run once)
// ---------------------------------------------------------------------------
__global__ void conv_bf16_kernel(const float* __restrict__ X,
                                 unsigned short* __restrict__ Xb, int n) {
    int i = blockIdx.x * 256 + threadIdx.x;
    if (i < n) Xb[i] = f2bf(X[i]);
}

// W[k][n] fp32 -> Wt[n][k] bf16  (pre-transpose so GEMM B-operand rows are
// contiguous WMMA B-fragments)
__global__ void transpose_conv_kernel(const float* __restrict__ W,
                                      unsigned short* __restrict__ Wt,
                                      int K, int N) {
    int n = blockIdx.x * 256 + threadIdx.x;
    int k = blockIdx.y;
    if (n < N) Wt[(size_t)n * K + k] = f2bf(W[(size_t)k * N + n]);
}

// ---------------------------------------------------------------------------
// GEMM: C[M,N] = A[M,K] (bf16) * Bt[N,K]^T (bf16), fp32 acc via WMMA.
// Block tile 128x128, 8 waves (4x2), each wave 32x64 via 2x4 WMMA tiles.
// Double-buffered LDS staging via global_load_async_to_lds_b128: the async
// DMA for tile i+1 runs concurrently with the 8 WMMAs on tile i.
// ---------------------------------------------------------------------------
__device__ __forceinline__ void stage_tile(const unsigned short* Ab,
                                           const unsigned short* Bb,
                                           int K, int k0,
                                           unsigned ldsA, unsigned ldsB, int tid) {
#pragma unroll
    for (int u = 0; u < 2; ++u) {
        int c = tid + u * 256;          // 0..511: 128 rows x 4 x 16B chunks
        int row = c >> 2;
        int part = c & 3;
        unsigned goff = (unsigned)(row * K + k0) * 2u + (unsigned)part * 16u;
        unsigned la = ldsA + (unsigned)(row * 64 + part * 16);
        unsigned lb = ldsB + (unsigned)(row * 64 + part * 16);
        asm volatile("global_load_async_to_lds_b128 %0, %1, %2 offset:0"
                     :: "v"(la), "v"(goff), "s"(Ab) : "memory");
        asm volatile("global_load_async_to_lds_b128 %0, %1, %2 offset:0"
                     :: "v"(lb), "v"(goff), "s"(Bb) : "memory");
    }
}

template <int C_BF16>
__global__ __launch_bounds__(256, 1) void gemm_bf16t(
    const unsigned short* __restrict__ A,   // [M][K] bf16
    const unsigned short* __restrict__ Bt,  // [N][K] bf16
    void* __restrict__ Cout, int M, int N, int K) {
    __shared__ __align__(16) unsigned short As[2][128 * 32];
    __shared__ __align__(16) unsigned short Bs[2][128 * 32];

    const int tid  = threadIdx.x;
    const int lane = tid & 31;
    const int w    = tid >> 5;
    const int wm   = w >> 1;        // 0..3  (32-row slots)
    const int wn   = w & 1;         // 0..1  (64-col slots)
    const int m0   = blockIdx.y * 128;
    const int n0   = blockIdx.x * 128;
    const int lr   = lane & 15;
    const int hi   = lane >> 4;

    f32x8 acc[2][4];
#pragma unroll
    for (int i = 0; i < 2; ++i)
#pragma unroll
        for (int j = 0; j < 4; ++j)
#pragma unroll
            for (int r = 0; r < 8; ++r) acc[i][j][r] = 0.0f;

    const unsigned short* Ab = A  + (size_t)m0 * K;
    const unsigned short* Bb = Bt + (size_t)n0 * K;
    const unsigned ldsA0 = (unsigned)(uintptr_t)(&As[0][0]);
    const unsigned ldsA1 = (unsigned)(uintptr_t)(&As[1][0]);
    const unsigned ldsB0 = (unsigned)(uintptr_t)(&Bs[0][0]);
    const unsigned ldsB1 = (unsigned)(uintptr_t)(&Bs[1][0]);

    const int nk = K >> 5;           // K/32 steps
    stage_tile(Ab, Bb, K, 0, ldsA0, ldsB0, tid);
    asm volatile("s_wait_asynccnt 0" ::: "memory");
    __syncthreads();

    for (int i = 0; i < nk; ++i) {
        const int cur = i & 1;
        if (i + 1 < nk)              // prefetch next tile into the other buffer
            stage_tile(Ab, Bb, K, (i + 1) << 5,
                       cur ? ldsA0 : ldsA1, cur ? ldsB0 : ldsB1, tid);

        Frag16 a[2], b[4];
#pragma unroll
        for (int ii = 0; ii < 2; ++ii) {
            const unsigned short* ap = &As[cur][(wm * 32 + ii * 16 + lr) * 32 + hi * 8];
            a[ii].q[0] = *(const u32x4*)(ap);
            a[ii].q[1] = *(const u32x4*)(ap + 16);
        }
#pragma unroll
        for (int j = 0; j < 4; ++j) {
            const unsigned short* bp = &Bs[cur][(wn * 64 + j * 16 + lr) * 32 + hi * 8];
            b[j].q[0] = *(const u32x4*)(bp);
            b[j].q[1] = *(const u32x4*)(bp + 16);
        }
#pragma unroll
        for (int ii = 0; ii < 2; ++ii)
#pragma unroll
            for (int j = 0; j < 4; ++j)
                acc[ii][j] = __builtin_amdgcn_wmma_f32_16x16x32_bf16(
                    false, a[ii].v, false, b[j].v, (short)0, acc[ii][j], false, false);

        asm volatile("s_wait_dscnt 0" ::: "memory");     // my LDS reads done
        asm volatile("s_wait_asynccnt 0" ::: "memory");  // next tile arrived
        __syncthreads();
    }

#pragma unroll
    for (int i = 0; i < 2; ++i)
#pragma unroll
        for (int j = 0; j < 4; ++j)
#pragma unroll
            for (int r = 0; r < 8; ++r) {
                int row = m0 + wm * 32 + i * 16 + r + 8 * hi;
                int col = n0 + wn * 64 + j * 16 + lr;
                if (C_BF16)
                    ((unsigned short*)Cout)[(size_t)row * N + col] = f2bf(acc[i][j][r]);
                else
                    ((float*)Cout)[(size_t)row * N + col] = acc[i][j][r];
            }
}

// ---------------------------------------------------------------------------
// Scaled RoPE (Llama-3 style) cos/sin on the fly
// ---------------------------------------------------------------------------
__device__ __forceinline__ void rope_cs(int pos, int i, float& c, float& s) {
    float inv = __powf(500000.0f, -(float)(2 * i) / 64.0f);
    const float old_ctx = 8192.0f, low_f = 1.0f, high_f = 4.0f, sf = 8.0f;
    float wl = 6.283185307179586f / inv;
    float f;
    if (wl < old_ctx / high_f) {
        f = inv;
    } else if (wl > old_ctx / low_f) {
        f = inv / sf;
    } else {
        float smooth = (old_ctx / wl - low_f) / (high_f - low_f);
        f = (1.0f - smooth) * inv / sf + smooth * inv;
    }
    float ang = (float)pos * f;
    sincosf(ang, &s, &c);
}

// Q also absorbs the 1/sqrt(HEAD_DIM) softmax scale (power of two: exact bf16)
__global__ void rope_q_kernel(const unsigned short* __restrict__ Y,
                              const int* __restrict__ pos,
                              unsigned short* __restrict__ Qb) {
    int idx = blockIdx.x * blockDim.x + threadIdx.x;
    if (idx >= SEQ * NH * 32) return;
    int s   = idx >> 10;
    int rem = idx & 1023;
    int h   = rem >> 5;
    int i   = rem & 31;
    float xr = bf2f(Y[(size_t)s * DMODEL + h * 64 + 2 * i]);
    float xi = bf2f(Y[(size_t)s * DMODEL + h * 64 + 2 * i + 1]);
    float c, sn; rope_cs(pos[s], i, c, sn);
    size_t o = ((size_t)h * SEQ + s) * 64 + 2 * i;
    Qb[o]     = f2bf((xr * c - xi * sn) * 0.125f);
    Qb[o + 1] = f2bf((xr * sn + xi * c) * 0.125f);
}

__global__ void rope_k_kernel(const unsigned short* __restrict__ Y,
                              const int* __restrict__ pos,
                              unsigned short* __restrict__ Kb) {
    int idx = blockIdx.x * blockDim.x + threadIdx.x;
    if (idx >= SEQ * NKV * 32) return;
    int s   = idx >> 8;
    int rem = idx & 255;
    int h   = rem >> 5;
    int i   = rem & 31;
    float xr = bf2f(Y[(size_t)s * (NKV * 64) + h * 64 + 2 * i]);
    float xi = bf2f(Y[(size_t)s * (NKV * 64) + h * 64 + 2 * i + 1]);
    float c, sn; rope_cs(pos[s], i, c, sn);
    size_t o = ((size_t)h * SP + s) * 64 + 2 * i;
    Kb[o]     = f2bf(xr * c - xi * sn);
    Kb[o + 1] = f2bf(xr * sn + xi * c);
}

__global__ void conv_v_kernel(const unsigned short* __restrict__ Y,
                              unsigned short* __restrict__ Vt) {
    int idx = blockIdx.x * blockDim.x + threadIdx.x;
    if (idx >= SEQ * NKV * 64) return;
    int s   = idx >> 9;
    int rem = idx & 511;
    int kv  = rem >> 6;
    int d   = rem & 63;
    Vt[((size_t)kv * 64 + d) * SP + s] = Y[(size_t)s * (NKV * 64) + kv * 64 + d];
}

// ---------------------------------------------------------------------------
// Flash attention: one wave owns a 16-row Q tile of one head, streams 32-key
// blocks causally. QK^T and PV both via v_wmma_f32_16x16x32_bf16.
// Softmax reductions run on the VALU pipe via DPP16 butterflies.
// ---------------------------------------------------------------------------
__global__ __launch_bounds__(256, 1) void flash_attn(
    const unsigned short* __restrict__ Qb,   // [NH][SEQ][64] bf16 (rope'd, pre-scaled)
    const unsigned short* __restrict__ Kb,   // [NKV][SP][64] bf16 (rope'd, pad=0)
    const unsigned short* __restrict__ Vt,   // [NKV][64][SP] bf16 (pad=0)
    const int* __restrict__ seg,             // attention_mask [SEQ]
    unsigned short* __restrict__ O) {        // [SEQ][DMODEL] bf16
    __shared__ __align__(16) unsigned short Plds[8][16 * 32];

    const int tid  = threadIdx.x;
    const int lane = tid & 31;
    const int w    = tid >> 5;
    const int lr   = lane & 15;
    const int hi   = lane >> 4;
    const int h    = blockIdx.x;
    const int kv   = h >> 2;                 // 4 Q-heads per KV head
    const int q0   = (blockIdx.y * 8 + w) * 16;

    // Q A-fragments (16x64 = two K=32 chunks)
    Frag16 aq[2];
    {
        const unsigned short* qrow = Qb + ((size_t)h * SEQ + q0 + lr) * HEAD_DIM;
#pragma unroll
        for (int dc = 0; dc < 2; ++dc) {
            const unsigned short* p = qrow + dc * 32 + hi * 8;
            aq[dc].q[0] = *(const u32x4*)(p);
            aq[dc].q[1] = *(const u32x4*)(p + 16);
        }
    }

    int segq[8];
#pragma unroll
    for (int r = 0; r < 8; ++r) segq[r] = seg[q0 + r + 8 * hi];

    float mrow[8], lrow[8];
    f32x8 oacc[4];
#pragma unroll
    for (int r = 0; r < 8; ++r) { mrow[r] = -1e30f; lrow[r] = 0.0f; }
#pragma unroll
    for (int t = 0; t < 4; ++t)
#pragma unroll
        for (int r = 0; r < 8; ++r) oacc[t][r] = 0.0f;

    const int qmax = q0 + 15;

    for (int kb = 0; kb <= qmax; kb += 32) {
        // --- scores: two 16x16 tiles (key cols kb..+15, kb+16..+31) ---
        // load all four K B-fragments up front so the loads can overlap
        Frag16 bk[2][2];    // [tile][dc]
        const unsigned short* k0row = Kb + ((size_t)kv * SP + kb + lr) * HEAD_DIM;
        const unsigned short* k1row = k0row + 16 * HEAD_DIM;
#pragma unroll
        for (int dc = 0; dc < 2; ++dc) {
            const unsigned short* p0 = k0row + dc * 32 + hi * 8;
            bk[0][dc].q[0] = *(const u32x4*)(p0);
            bk[0][dc].q[1] = *(const u32x4*)(p0 + 16);
            const unsigned short* p1 = k1row + dc * 32 + hi * 8;
            bk[1][dc].q[0] = *(const u32x4*)(p1);
            bk[1][dc].q[1] = *(const u32x4*)(p1 + 16);
        }
        f32x8 s0, s1;
#pragma unroll
        for (int r = 0; r < 8; ++r) { s0[r] = 0.0f; s1[r] = 0.0f; }
#pragma unroll
        for (int dc = 0; dc < 2; ++dc) {
            s0 = __builtin_amdgcn_wmma_f32_16x16x32_bf16(false, aq[dc].v, false, bk[0][dc].v,
                                                         (short)0, s0, false, false);
            s1 = __builtin_amdgcn_wmma_f32_16x16x32_bf16(false, aq[dc].v, false, bk[1][dc].v,
                                                         (short)0, s1, false, false);
        }

        const int kp0 = kb + lr;
        const int kp1 = kp0 + 16;
        const int sk0 = (kp0 < SEQ) ? seg[kp0] : -1;
        const int sk1 = (kp1 < SEQ) ? seg[kp1] : -1;

        // --- online softmax (row = r + 8*hi, col = lane&15 [+16]) ---
#pragma unroll
        for (int r = 0; r < 8; ++r) {
            const int qp = q0 + r + 8 * hi;
            float x0 = s0[r];
            float x1 = s1[r];
            if (kp0 > qp || sk0 != segq[r]) x0 = -1e30f;
            if (kp1 > qp || sk1 != segq[r]) x1 = -1e30f;
            float mx = grp16_max(fmaxf(x0, x1));
            float mn = fmaxf(mrow[r], mx);
            float p0 = __expf(x0 - mn);
            float p1 = __expf(x1 - mn);
            float alpha = __expf(mrow[r] - mn);
            float ps = grp16_sum(p0 + p1);
            lrow[r] = alpha * lrow[r] + ps;
            mrow[r] = mn;
#pragma unroll
            for (int t = 0; t < 4; ++t) oacc[t][r] *= alpha;
            const int prow = r + 8 * hi;
            Plds[w][prow * 32 + lr]      = f2bf(p0);
            Plds[w][prow * 32 + lr + 16] = f2bf(p1);
        }

        // per-wave LDS RAW fence (C-layout -> A-layout re-shuffle)
        asm volatile("s_wait_dscnt 0" ::: "memory");

        Frag16 ap;
        {
            const unsigned short* pp = &Plds[w][lr * 32 + hi * 8];
            ap.q[0] = *(const u32x4*)(pp);
            ap.q[1] = *(const u32x4*)(pp + 16);
        }
        // --- PV: 4 d-tiles of 16 (B^T rows are Vt rows: contiguous in s) ---
        Frag16 bv[4];
#pragma unroll
        for (int t = 0; t < 4; ++t) {
            const unsigned short* vp =
                Vt + ((size_t)kv * HEAD_DIM + t * 16 + lr) * SP + kb + hi * 8;
            bv[t].q[0] = *(const u32x4*)(vp);
            bv[t].q[1] = *(const u32x4*)(vp + 16);
        }
#pragma unroll
        for (int t = 0; t < 4; ++t)
            oacc[t] = __builtin_amdgcn_wmma_f32_16x16x32_bf16(false, ap.v, false, bv[t].v,
                                                              (short)0, oacc[t], false, false);
    }

#pragma unroll
    for (int t = 0; t < 4; ++t)
#pragma unroll
        for (int r = 0; r < 8; ++r) {
            int row = q0 + r + 8 * hi;
            int col = h * HEAD_DIM + t * 16 + lr;
            O[(size_t)row * DMODEL + col] = f2bf(oacc[t][r] / lrow[r]);
        }
}

// ---------------------------------------------------------------------------
extern "C" void kernel_launch(void* const* d_in, const int* in_sizes, int n_in,
                              void* d_out, int out_size, void* d_ws, size_t ws_size,
                              hipStream_t stream) {
    (void)in_sizes; (void)n_in; (void)out_size; (void)ws_size;
    const float* X    = (const float*)d_in[0];
    const int*   mask = (const int*)d_in[1];
    const int*   pos  = (const int*)d_in[2];
    const float* wq   = (const float*)d_in[3];
    const float* wk   = (const float*)d_in[4];
    const float* wv   = (const float*)d_in[5];
    const float* wo   = (const float*)d_in[6];
    float* out = (float*)d_out;

    char* ws = (char*)d_ws;
    size_t off = 0;
    auto alloc = [&](size_t bytes) -> char* {
        char* p = ws + off;
        off += (bytes + 255) & ~(size_t)255;
        return p;
    };
    unsigned short* Xbf = (unsigned short*)alloc((size_t)SEQ * DMODEL * 2); // reused as attn-out bf16
    unsigned short* WqT = (unsigned short*)alloc((size_t)DMODEL * DMODEL * 2);
    unsigned short* WkT = (unsigned short*)alloc((size_t)(NKV * 64) * DMODEL * 2);
    unsigned short* WvT = (unsigned short*)alloc((size_t)(NKV * 64) * DMODEL * 2);
    unsigned short* WoT = (unsigned short*)alloc((size_t)DMODEL * DMODEL * 2);
    unsigned short* Yqb = (unsigned short*)alloc((size_t)SEQ * DMODEL * 2);
    unsigned short* Ykb = (unsigned short*)alloc((size_t)SEQ * NKV * 64 * 2);
    unsigned short* Yvb = (unsigned short*)alloc((size_t)SEQ * NKV * 64 * 2);
    unsigned short* Qb  = (unsigned short*)alloc((size_t)NH * SEQ * 64 * 2);
    unsigned short* Kb  = (unsigned short*)alloc((size_t)NKV * SP * 64 * 2);
    unsigned short* Vt  = (unsigned short*)alloc((size_t)NKV * 64 * SP * 2);

    dim3 blk(256);

    // one-time bf16 conversion / weight pre-transpose
    conv_bf16_kernel<<<(SEQ * DMODEL + 255) / 256, blk, 0, stream>>>(X, Xbf, SEQ * DMODEL);
    transpose_conv_kernel<<<dim3(DMODEL / 256, DMODEL), blk, 0, stream>>>(wq, WqT, DMODEL, DMODEL);
    transpose_conv_kernel<<<dim3((NKV * 64) / 256, DMODEL), blk, 0, stream>>>(wk, WkT, DMODEL, NKV * 64);
    transpose_conv_kernel<<<dim3((NKV * 64) / 256, DMODEL), blk, 0, stream>>>(wv, WvT, DMODEL, NKV * 64);
    transpose_conv_kernel<<<dim3(DMODEL / 256, DMODEL), blk, 0, stream>>>(wo, WoT, DMODEL, DMODEL);

    // QKV projections (bf16 WMMA, double-buffered async-LDS staging)
    gemm_bf16t<1><<<dim3(DMODEL / 128, SEQ / 128), blk, 0, stream>>>(Xbf, WqT, Yqb, SEQ, DMODEL, DMODEL);
    gemm_bf16t<1><<<dim3((NKV * 64) / 128, SEQ / 128), blk, 0, stream>>>(Xbf, WkT, Ykb, SEQ, NKV * 64, DMODEL);
    gemm_bf16t<1><<<dim3((NKV * 64) / 128, SEQ / 128), blk, 0, stream>>>(Xbf, WvT, Yvb, SEQ, NKV * 64, DMODEL);

    // zero-fill K/V (covers the causal-overrun pad rows/cols)
    hipMemsetAsync(Kb, 0, (size_t)NKV * SP * 64 * 2, stream);
    hipMemsetAsync(Vt, 0, (size_t)NKV * 64 * SP * 2, stream);

    rope_q_kernel<<<(SEQ * NH * 32 + 255) / 256, blk, 0, stream>>>(Yqb, pos, Qb);
    rope_k_kernel<<<(SEQ * NKV * 32 + 255) / 256, blk, 0, stream>>>(Ykb, pos, Kb);
    conv_v_kernel<<<(SEQ * NKV * 64 + 255) / 256, blk, 0, stream>>>(Yvb, Vt);

    // flash attention -> bf16 attn-out (reuses Xbf, already consumed)
    flash_attn<<<dim3(NH, SEQ / 128), blk, 0, stream>>>(Qb, Kb, Vt, mask, Xbf);

    // output projection -> fp32 d_out
    gemm_bf16t<0><<<dim3(DMODEL / 128, SEQ / 128), blk, 0, stream>>>(Xbf, WoT, out, SEQ, DMODEL, DMODEL);
}
